// MultiScaleRetention_67662914781768
// MI455X (gfx1250) — matrix operand order repaired and checked by reading, not compiled, not run on hardware
//
#include <hip/hip_runtime.h>
#include <hip/hip_bf16.h>

// ---------------------------------------------------------------------------
// MultiScaleRetention forward for MI455X (gfx1250, wave32, WMMA bf16)
// B=2, S=1024, H=2048, nheads=16, d=128
// Pipelined: LDS double-buffering over GLOBAL_LOAD_ASYNC_TO_LDS_B128 with
// partial s_wait_asynccnt (async loads complete in order, ISA 4.1).
// ---------------------------------------------------------------------------

typedef __attribute__((ext_vector_type(16))) __bf16 v16bf;
typedef __attribute__((ext_vector_type(8)))  __bf16 v8bf;
typedef __attribute__((ext_vector_type(8)))  float  v8f;

#define NB    2
#define NS    1024
#define NH    2048
#define NHEAD 16
#define ND    128
#define MROWS (NB * NS)   // 2048

// ---------------------------------------------------------------------------
// Async global -> LDS copy (CDNA5 GLOBAL_LOAD_ASYNC_TO_LDS_B128, ASYNCcnt).
// Builtin signature (probe-verified): (as1 int4*, as3 int4*, imm off, imm cpol)
// ---------------------------------------------------------------------------

#if defined(__has_builtin)
#if __has_builtin(__builtin_amdgcn_global_load_async_to_lds_b128)
#define HAS_ASYNC_LDS 1
#endif
#if __has_builtin(__builtin_amdgcn_s_wait_asynccnt)
#define HAS_WAIT_ASYNC 1
#endif
#endif

typedef int async_v4i __attribute__((vector_size(16)));
typedef __attribute__((address_space(1))) async_v4i* g_v4i_ptr;
typedef __attribute__((address_space(3))) async_v4i* l_v4i_ptr;

__device__ __forceinline__ void cp_g2l_b128(void* lds, const void* gm) {
#if defined(HAS_ASYNC_LDS)
    __builtin_amdgcn_global_load_async_to_lds_b128(
        (g_v4i_ptr)(void*)gm, (l_v4i_ptr)lds, 0, 0);
#else
    *(uint4*)lds = *(const uint4*)gm;
#endif
}

#if defined(HAS_ASYNC_LDS)
#if defined(HAS_WAIT_ASYNC)
#define ASYNC_WAIT(n) __builtin_amdgcn_s_wait_asynccnt(n)
#else
#define ASYNC_WAIT(n) asm volatile("s_wait_asynccnt %0" ::"i"(n) : "memory")
#endif
#else
#define ASYNC_WAIT(n) ((void)0)
#endif

// Stage an R x C bf16 tile (C multiple of 8) from global (row stride
// srcStride) into LDS row-major [R][C]. Issues exactly R*C/8/256 async
// instructions per wave (256-thread blocks).
template <int R, int C>
__device__ __forceinline__ void stage_tile(__bf16* dst, const __bf16* src,
                                           size_t srcStride) {
    constexpr int CH = C / 8;
#pragma unroll
    for (int idx = threadIdx.x; idx < R * CH; idx += 256) {
        int r = idx / CH;
        int c = (idx % CH) * 8;
        cp_g2l_b128(&dst[r * C + c], &src[(size_t)r * srcStride + c]);
    }
}

// ---------------------------------------------------------------------------
// WMMA fragment helpers (layouts per CDNA5 ISA 7.12.2)
// ---------------------------------------------------------------------------

// A-matrix 16x32 bf16: lane&15 = M row; lanes 0-15 hold K[0:8)+K[16:24),
// lanes 16-31 hold K[8:16)+K[24:32). Two 16B contiguous LDS loads per lane.
__device__ __forceinline__ v16bf load_a_frag(const __bf16* base, int ld) {
    int lane = threadIdx.x & 31;
    int m    = lane & 15;
    int kb   = (lane >> 4) * 8;
    const __bf16* p = base + m * ld;
    v8bf g0 = *(const v8bf*)(p + kb);
    v8bf g1 = *(const v8bf*)(p + kb + 16);
    v16bf f;
#pragma unroll
    for (int i = 0; i < 8; ++i) { f[i] = g0[i]; f[i + 8] = g1[i]; }
    return f;
}

// B-matrix 32x16 bf16 staged transposed in LDS as [N][K]: lane&15 = N col;
// lanes 0-15 read K[0:8)+K[16:24), lanes 16-31 read K[8:16)+K[24:32).
__device__ __forceinline__ v16bf load_bt_frag(const __bf16* base, int ld) {
    int lane = threadIdx.x & 31;
    int n    = lane & 15;
    int kb   = (lane >> 4) * 8;
    const __bf16* p = base + n * ld + kb;
    v8bf g0 = *(const v8bf*)(p);
    v8bf g1 = *(const v8bf*)(p + 16);
    v16bf f;
#pragma unroll
    for (int i = 0; i < 8; ++i) { f[i] = g0[i]; f[i + 8] = g1[i]; }
    return f;
}

__device__ __forceinline__ v8f wmma_bf16(v16bf a, v16bf b, v8f c) {
    return __builtin_amdgcn_wmma_f32_16x16x32_bf16(false, a, false, b,
                                                   (short)0, c, false, false);
}

// C/D 16x16 f32 layout: VGPR i -> row = (lane>>4)*8 + i, col = lane&15
__device__ __forceinline__ void store_c_tile(float* C, int ldc, v8f acc) {
    int lane = threadIdx.x & 31;
    int col  = lane & 15;
    int rb   = (lane >> 4) * 8;
#pragma unroll
    for (int i = 0; i < 8; ++i) C[(size_t)(rb + i) * ldc + col] = acc[i];
}

// ---------------------------------------------------------------------------
// Conversion kernels
// ---------------------------------------------------------------------------

__global__ void cvt_bf16(const float* __restrict__ src, __bf16* __restrict__ dst) {
    size_t i = (size_t)blockIdx.x * blockDim.x + threadIdx.x;
    dst[i] = (__bf16)src[i];
}

// W_{Q,K,V} (nheads, H, d) -> Bt layout [N=n*128+dd][K=h]
__global__ void cvt_w_heads(const float* __restrict__ W, __bf16* __restrict__ Wt) {
    size_t idx = (size_t)blockIdx.x * blockDim.x + threadIdx.x;
    int col = (int)(idx >> 11);          // 0..2047  (n*128+dd)
    int h   = (int)(idx & 2047);
    int n   = col >> 7;
    int dd  = col & 127;
    Wt[idx] = (__bf16)W[((size_t)n * NH + h) * ND + dd];
}

// W (H,H) row-major -> Bt layout [N][K] (transpose)
__global__ void cvt_w_transpose(const float* __restrict__ W, __bf16* __restrict__ Wt) {
    size_t idx = (size_t)blockIdx.x * blockDim.x + threadIdx.x;
    int ncol = (int)(idx >> 11);
    int k    = (int)(idx & 2047);
    Wt[idx] = (__bf16)W[(size_t)k * NH + ncol];
}

// ---------------------------------------------------------------------------
// GEMM: C[M,N] (f32) = A[M,K] (bf16, row-major) x Bt[N,K] (bf16, N-major)
// 128x128 block tile, BK=32, 256 threads = 8 waves, wave tile 64x32.
// Double-buffered async staging: tile k+1 in flight behind tile k's WMMAs;
// per tile each wave issues 4 async instrs, so s_wait_asynccnt 4 retires
// exactly the older tile.
// ---------------------------------------------------------------------------

#define BM 128
#define BN 128
#define BK 32

__global__ __launch_bounds__(256) void gemm_bf16(const __bf16* __restrict__ A,
                                                 const __bf16* __restrict__ Bt,
                                                 float* __restrict__ C,
                                                 int M, int N, int K) {
    __shared__ __attribute__((aligned(16))) __bf16 sA[2][BM * BK];
    __shared__ __attribute__((aligned(16))) __bf16 sB[2][BN * BK];
    int m0 = blockIdx.y * BM;
    int n0 = blockIdx.x * BN;
    int w  = threadIdx.x >> 5;
    int wm = (w >> 2) * 64;   // wave M offset: 0 / 64
    int wn = (w & 3) * 32;    // wave N offset: 0..96

    const __bf16* Ab = A  + (size_t)m0 * K;
    const __bf16* Bb = Bt + (size_t)n0 * K;

    v8f zero = {};
    v8f acc[4][2];
#pragma unroll
    for (int i = 0; i < 4; ++i)
#pragma unroll
        for (int j = 0; j < 2; ++j) acc[i][j] = zero;

    // Prologue: stage tile 0 into buffer 0.
    stage_tile<BM, BK>(sA[0], Ab, K);
    stage_tile<BN, BK>(sB[0], Bb, K);

    int nk = K / BK;
    for (int kt = 0; kt < nk; ++kt) {
        int buf = kt & 1;
        if (kt + 1 < nk) {
            // Issue next tile into the other buffer (its last readers
            // finished before the end-of-iteration barrier of kt-1).
            size_t k1 = (size_t)(kt + 1) * BK;
            stage_tile<BM, BK>(sA[buf ^ 1], Ab + k1, K);
            stage_tile<BN, BK>(sB[buf ^ 1], Bb + k1, K);
            // Cache prefetch two tiles ahead.
            if (kt + 2 < nk) {
                int r = threadIdx.x >> 1;
                __builtin_prefetch(&Ab[(size_t)r * K + k1 + BK], 0, 3);
                __builtin_prefetch(&Bb[(size_t)r * K + k1 + BK], 0, 3);
            }
            ASYNC_WAIT(4);   // retire tile kt; tile kt+1 stays in flight
        } else {
            ASYNC_WAIT(0);
        }
        __syncthreads();

        v16bf bfr[2];
#pragma unroll
        for (int j = 0; j < 2; ++j)
            bfr[j] = load_bt_frag(&sB[buf][(wn + j * 16) * BK], BK);
#pragma unroll
        for (int i = 0; i < 4; ++i) {
            v16bf afr = load_a_frag(&sA[buf][(wm + i * 16) * BK], BK);
#pragma unroll
            for (int j = 0; j < 2; ++j)
                acc[i][j] = wmma_bf16(afr, bfr[j], acc[i][j]);
        }
        __syncthreads();
    }

#pragma unroll
    for (int i = 0; i < 4; ++i)
#pragma unroll
        for (int j = 0; j < 2; ++j)
            store_c_tile(&C[(size_t)(m0 + wm + i * 16) * N + n0 + wn + j * 16],
                         N, acc[i][j]);
}

// ---------------------------------------------------------------------------
// xPos epilogue: apply rotary + scale (or 1/scale), emit bf16 in per-head
// layout Q[(b*16+n)*1024 + s][dd].
// ---------------------------------------------------------------------------

__global__ void xpos_qk(const float* __restrict__ P, __bf16* __restrict__ dst,
                        int invScale) {
    size_t idx = (size_t)blockIdx.x * blockDim.x + threadIdx.x; // one per pair
    int pair = (int)(idx & 1023);
    int row  = (int)(idx >> 10);          // b*1024+s
    int b = row >> 10, s = row & 1023;
    int col = pair * 2;                   // even column in [0,2048)
    int n = col >> 7, dd = col & 127;
    int hi = (dd >> 1);                   // half index 0..63

    float x0 = P[(size_t)row * NH + col];
    float x1 = P[(size_t)row * NH + col + 1];

    float xscale  = (2.f * hi + 0.4f * 128.f) / (1.4f * 128.f);
    float scale   = __powf(xscale, (float)s / 512.f);
    if (invScale) scale = 1.f / scale;
    float invfreq = __powf(10000.f, -(float)hi / 64.f);
    float sn, cs;
    __sincosf((float)s * invfreq, &sn, &cs);
    float c  = cs * scale;
    float ss = sn * scale;
    float o0 = x0 * c - x1 * ss;          // x*cos + rot(x)*sin, rot = (-x1, x0)
    float o1 = x1 * c + x0 * ss;

    size_t o = (((size_t)(b * NHEAD + n) * NS + s) * ND) + dd;
    dst[o]     = (__bf16)o0;
    dst[o + 1] = (__bf16)o1;
}

// V (f32, [b*1024+s][n*128+dd]) -> Vt bf16 per head [dd][s]
__global__ void v_transpose(const float* __restrict__ Vf, __bf16* __restrict__ Vt) {
    size_t idx = (size_t)blockIdx.x * blockDim.x + threadIdx.x;
    int row = (int)(idx >> 11);
    int col = (int)(idx & 2047);
    int b = row >> 10, s = row & 1023;
    int n = col >> 7, dd = col & 127;
    Vt[(((size_t)(b * NHEAD + n) * ND + dd) * NS) + s] = (__bf16)Vf[idx];
}

// ---------------------------------------------------------------------------
// Retention: per (b,head) and 64-row Q block, causal chunked loop:
//   P = Q Kt  (64x64, f32 acc)  ->  * gamma^(s-t)  ->  bf16  ->  Y += P V
// Double-buffered K/Vt chunks: 8 async instrs per chunk per wave, so
// s_wait_asynccnt 8 retires chunk tc (and sQ on the first iteration) while
// chunk tc+1 stays in flight.
// ---------------------------------------------------------------------------

__global__ __launch_bounds__(256) void retention_kernel(
    const __bf16* __restrict__ Qbf, const __bf16* __restrict__ Kbf,
    const __bf16* __restrict__ Vtbf, float* __restrict__ Y) {
    __shared__ __attribute__((aligned(16))) __bf16 sQ[64 * 128];
    __shared__ __attribute__((aligned(16))) __bf16 sK[2][64 * 128];  // [t][dd]
    __shared__ __attribute__((aligned(16))) __bf16 sVt[2][128 * 64]; // [dd][t]
    __shared__ __attribute__((aligned(16))) __bf16 sP[64 * 64];

    int bn   = blockIdx.x;            // 0..31
    int b    = bn >> 4, n = bn & 15;
    int sblk = blockIdx.y;            // 0..15
    int s0   = sblk * 64;

    const __bf16* Qh  = Qbf  + (size_t)bn * NS * ND;
    const __bf16* Kh  = Kbf  + (size_t)bn * NS * ND;
    const __bf16* Vth = Vtbf + (size_t)bn * ND * NS;

    // gammas = 1 - exp(linspace(log(1/32), log(1/512), 16))
    const float la = -3.46573590f, lb = -6.23832463f;
    float gamma = 1.f - __expf(la + (lb - la) * ((float)n / 15.f));
    float l2g   = __log2f(gamma);

    int w    = threadIdx.x >> 5;
    int lane = threadIdx.x & 31;
    int wm   = (w >> 2) * 32;   // rows (scores M and Y M)
    int wn   = (w & 3) * 16;    // scores N within 64
    int wn2  = (w & 3) * 32;    // Y N within 128

    // Prologue: Q block + chunk 0 (async, in order).
    stage_tile<64, 128>(sQ, Qh + (size_t)s0 * ND, ND);
    stage_tile<64, 128>(sK[0], Kh, ND);
    stage_tile<128, 64>(sVt[0], Vth, NS);

    v8f zero = {};
    v8f yacc[2][2];
#pragma unroll
    for (int i = 0; i < 2; ++i)
#pragma unroll
        for (int j = 0; j < 2; ++j) yacc[i][j] = zero;

    for (int tc = 0; tc <= sblk; ++tc) {
        int t0  = tc * 64;
        int buf = tc & 1;
        if (tc + 1 <= sblk) {
            size_t t1 = (size_t)(tc + 1) * 64;
            stage_tile<64, 128>(sK[buf ^ 1], Kh + t1 * ND, ND);
            stage_tile<128, 64>(sVt[buf ^ 1], Vth + t1, NS);
            ASYNC_WAIT(8);   // retire chunk tc (+sQ at tc==0)
        } else {
            ASYNC_WAIT(0);
        }
        __syncthreads();

        // Scores: 2 M-tiles x 1 N-tile per wave, K=128 in 4 steps of 32.
        v8f p[2] = {zero, zero};
#pragma unroll
        for (int kt = 0; kt < 4; ++kt) {
            v16bf bfr = load_bt_frag(&sK[buf][wn * 128 + kt * 32], 128);
#pragma unroll
            for (int i = 0; i < 2; ++i) {
                v16bf afr = load_a_frag(&sQ[(wm + i * 16) * 128 + kt * 32], 128);
                p[i] = wmma_bf16(afr, bfr, p[i]);
            }
        }

        // Decay + causal mask; spill to LDS as bf16 A-matrix for P@V.
        int col = lane & 15;
        int rb  = (lane >> 4) * 8;
        int t   = t0 + wn + col;
#pragma unroll
        for (int i = 0; i < 2; ++i) {
#pragma unroll
            for (int r = 0; r < 8; ++r) {
                int s     = s0 + wm + i * 16 + rb + r;
                int delta = s - t;
                float v = (delta >= 0) ? p[i][r] * exp2f(l2g * (float)delta)
                                       : 0.f;
                sP[(wm + i * 16 + rb + r) * 64 + wn + col] = (__bf16)v;
            }
        }
        __syncthreads();

        // Y += P @ V : wave tile 32x32, K=64 in 2 steps of 32.
#pragma unroll
        for (int kt = 0; kt < 2; ++kt) {
#pragma unroll
            for (int j = 0; j < 2; ++j) {
                v16bf bfr = load_bt_frag(&sVt[buf][(wn2 + j * 16) * 64 + kt * 32], 64);
#pragma unroll
                for (int i = 0; i < 2; ++i) {
                    v16bf afr = load_a_frag(&sP[(wm + i * 16) * 64 + kt * 32], 64);
                    yacc[i][j] = wmma_bf16(afr, bfr, yacc[i][j]);
                }
            }
        }
        __syncthreads();
    }

    // Write Y in (b, s, n, dd) layout -> flat (B,S,H).
    int col = lane & 15;
    int rb  = (lane >> 4) * 8;
#pragma unroll
    for (int i = 0; i < 2; ++i)
#pragma unroll
        for (int j = 0; j < 2; ++j)
#pragma unroll
            for (int r = 0; r < 8; ++r) {
                int s = s0 + wm + i * 16 + rb + r;
                int d = wn2 + j * 16 + col;
                Y[(((size_t)b * NS + s) * NHEAD + n) * ND + d] = yacc[i][j][r];
            }
}

// ---------------------------------------------------------------------------
// GroupNorm over d=128 per head (one wave per group) fused with SiLU gate.
// ---------------------------------------------------------------------------

__global__ __launch_bounds__(256) void gn_gate_kernel(
    const float* __restrict__ Y, const float* __restrict__ G,
    const float* __restrict__ gw, const float* __restrict__ gb,
    __bf16* __restrict__ Abf) {
    int gidx = blockIdx.x * 8 + (threadIdx.x >> 5);  // 0..32767
    int lane = threadIdx.x & 31;
    size_t flat = (size_t)gidx * ND + lane * 4;

    float4 v = *(const float4*)(Y + flat);
    float sum = v.x + v.y + v.z + v.w;
    float sq  = v.x * v.x + v.y * v.y + v.z * v.z + v.w * v.w;
#pragma unroll
    for (int off = 16; off; off >>= 1) {
        sum += __shfl_xor(sum, off, 32);
        sq  += __shfl_xor(sq,  off, 32);
    }
    float mu  = sum * (1.f / 128.f);
    float var = sq * (1.f / 128.f) - mu * mu;
    float inv = __frsqrt_rn(var + 1e-5f);

    int h0 = (gidx & 15) * ND + lane * 4;      // n*128 + dd
    float4 gv = *(const float4*)(G + flat);
    float4 w4 = *(const float4*)(gw + h0);
    float4 b4 = *(const float4*)(gb + h0);

    float yn0 = (v.x - mu) * inv * w4.x + b4.x;
    float yn1 = (v.y - mu) * inv * w4.y + b4.y;
    float yn2 = (v.z - mu) * inv * w4.z + b4.z;
    float yn3 = (v.w - mu) * inv * w4.w + b4.w;
    float g0 = gv.x / (1.f + __expf(-gv.x));
    float g1 = gv.y / (1.f + __expf(-gv.y));
    float g2 = gv.z / (1.f + __expf(-gv.z));
    float g3 = gv.w / (1.f + __expf(-gv.w));

    Abf[flat]     = (__bf16)(g0 * yn0);
    Abf[flat + 1] = (__bf16)(g1 * yn1);
    Abf[flat + 2] = (__bf16)(g2 * yn2);
    Abf[flat + 3] = (__bf16)(g3 * yn3);
}

// ---------------------------------------------------------------------------
// Host-side orchestration
// ---------------------------------------------------------------------------

extern "C" void kernel_launch(void* const* d_in, const int* in_sizes, int n_in,
                              void* d_out, int out_size, void* d_ws, size_t ws_size,
                              hipStream_t stream) {
    const float* X   = (const float*)d_in[0];
    const float* W_Q = (const float*)d_in[1];
    const float* W_K = (const float*)d_in[2];
    const float* W_V = (const float*)d_in[3];
    const float* W_G = (const float*)d_in[4];
    const float* W_O = (const float*)d_in[5];
    const float* gnw = (const float*)d_in[6];
    const float* gnb = (const float*)d_in[7];
    float* out = (float*)d_out;

    const size_t MB = 1024ull * 1024ull;
    char* ws = (char*)d_ws;
    // Phase-1 regions (freed after projection GEMMs, then reused)
    __bf16* Xbf = (__bf16*)(ws + 0 * MB);     // 8MB
    __bf16* Wqt = (__bf16*)(ws + 8 * MB);     // 8MB
    __bf16* Wkt = (__bf16*)(ws + 16 * MB);    // 8MB
    __bf16* Wvt = (__bf16*)(ws + 24 * MB);    // 8MB
    __bf16* Wgt = (__bf16*)(ws + 32 * MB);    // 8MB
    __bf16* Wot = (__bf16*)(ws + 40 * MB);    // 8MB (live until final GEMM)
    float*  Qf  = (float*)(ws + 48 * MB);     // 16MB
    float*  Kf  = (float*)(ws + 64 * MB);     // 16MB
    float*  Vf  = (float*)(ws + 80 * MB);     // 16MB
    float*  Gf  = (float*)(ws + 96 * MB);     // 16MB  -> total 112MB
    // Phase-2 aliases (safe by stream ordering)
    __bf16* Qbf  = (__bf16*)(ws + 0 * MB);    // over Xbf
    __bf16* Kbf  = (__bf16*)(ws + 8 * MB);    // over Wqt
    __bf16* Vtbf = (__bf16*)(ws + 16 * MB);   // over Wkt
    __bf16* Abf  = (__bf16*)(ws + 24 * MB);   // over Wvt
    float*  Yf   = (float*)(ws + 48 * MB);    // over Qf

    const int ELEMS = NH * NH;                // 2048*2048 = 4,194,304
    dim3 blk(256);
    dim3 cvtGrid(ELEMS / 256);

    // 1) bf16 conversions (B matrices pre-transposed to [N][K]).
    cvt_bf16<<<cvtGrid, blk, 0, stream>>>(X, Xbf);
    cvt_w_heads<<<cvtGrid, blk, 0, stream>>>(W_Q, Wqt);
    cvt_w_heads<<<cvtGrid, blk, 0, stream>>>(W_K, Wkt);
    cvt_w_heads<<<cvtGrid, blk, 0, stream>>>(W_V, Wvt);
    cvt_w_transpose<<<cvtGrid, blk, 0, stream>>>(W_G, Wgt);
    cvt_w_transpose<<<cvtGrid, blk, 0, stream>>>(W_O, Wot);

    // 2) Projection GEMMs (M=N=K=2048).
    dim3 gemmGrid(NH / BN, MROWS / BM);
    gemm_bf16<<<gemmGrid, blk, 0, stream>>>(Xbf, Wqt, Qf, MROWS, NH, NH);
    gemm_bf16<<<gemmGrid, blk, 0, stream>>>(Xbf, Wkt, Kf, MROWS, NH, NH);
    gemm_bf16<<<gemmGrid, blk, 0, stream>>>(Xbf, Wvt, Vf, MROWS, NH, NH);
    gemm_bf16<<<gemmGrid, blk, 0, stream>>>(Xbf, Wgt, Gf, MROWS, NH, NH);

    // 3) xPos epilogues + V transpose (bf16, per-head layouts).
    dim3 pairGrid(ELEMS / 2 / 256);
    xpos_qk<<<pairGrid, blk, 0, stream>>>(Qf, Qbf, 0);
    xpos_qk<<<pairGrid, blk, 0, stream>>>(Kf, Kbf, 1);
    v_transpose<<<cvtGrid, blk, 0, stream>>>(Vf, Vtbf);

    // 4) Causal retention (decay-weighted attention) -> Y (B,S,H) f32.
    dim3 retGrid(NB * NHEAD, NS / 64);
    retention_kernel<<<retGrid, blk, 0, stream>>>(Qbf, Kbf, Vtbf, Yf);

    // 5) GroupNorm + SiLU gate -> bf16 A.
    dim3 gnGrid(NB * NS * NHEAD / 8);
    gn_gate_kernel<<<gnGrid, blk, 0, stream>>>(Yf, Gf, gnw, gnb, Abf);

    // 6) Output GEMM -> d_out (f32, (B,S,H)).
    gemm_bf16<<<gemmGrid, blk, 0, stream>>>(Abf, Wot, out, MROWS, NH, NH);
}
